// NVFP4InferenceGroupedLinear_5093831213306
// MI455X (gfx1250) — matrix operand order repaired and checked by reading, not compiled, not run on hardware
//
#include <hip/hip_runtime.h>
#include <hip/hip_bf16.h>

typedef __attribute__((ext_vector_type(16))) _Float16 v16h;
typedef __attribute__((ext_vector_type(8)))  _Float16 v8h;
typedef __attribute__((ext_vector_type(8)))  float    v8f;
typedef __attribute__((ext_vector_type(4)))  float    v4f;

#define F8_TINY 0.015625f   /* 2^-6, smallest e4m3 normal */
#define F8_MAX  448.0f
#define FP4_MAX 6.0f

// ---------------------------------------------------------------------------
// Kernel 1: zero the per-group absmax accumulators in workspace.
// ---------------------------------------------------------------------------
__global__ void nvfp4_init_kernel(unsigned* __restrict__ amax, int G) {
    if ((int)threadIdx.x < G) amax[threadIdx.x] = 0u;
}

// ---------------------------------------------------------------------------
// Kernel 2: per-group absolute max (gs[g] = max|W[g,:,:]|).
// Positive-float bit patterns are monotonic as unsigned ints -> atomicMax.
// ---------------------------------------------------------------------------
__global__ void __launch_bounds__(256)
nvfp4_absmax_kernel(const float* __restrict__ w, unsigned* __restrict__ amax,
                    long long perGroup) {
    __shared__ float red[256];
    const long long g = blockIdx.y;
    const float* p = w + g * perGroup;
    float m = 0.0f;
    for (long long i = (long long)blockIdx.x * blockDim.x + threadIdx.x;
         i < perGroup; i += (long long)gridDim.x * blockDim.x)
        m = fmaxf(m, fabsf(p[i]));
    red[threadIdx.x] = m;
    __syncthreads();
    for (int s = 128; s > 0; s >>= 1) {
        if ((int)threadIdx.x < s)
            red[threadIdx.x] = fmaxf(red[threadIdx.x], red[threadIdx.x + s]);
        __syncthreads();
    }
    if (threadIdx.x == 0) atomicMax(&amax[g], __float_as_uint(red[0]));
}

// ---------------------------------------------------------------------------
// fp8 e4m3 round-to-nearest-even for v in [2^-6, 448] (positive, normal).
// ---------------------------------------------------------------------------
__device__ __forceinline__ float f8e4m3_rne(float v) {
    unsigned b   = __float_as_uint(v);
    unsigned exp = (b >> 23) & 0xFFu;
    unsigned man = b & 0x7FFFFFu;
    unsigned m3  = man >> 20;
    unsigned rem = man & 0xFFFFFu;
    if (rem > 0x80000u || (rem == 0x80000u && (m3 & 1u))) m3++;
    if (m3 == 8u) { m3 = 0u; exp++; }
    float r = __uint_as_float((exp << 23) | (m3 << 20));
    return fminf(r, F8_MAX);
}

// fp4 e2m1 RNE for x already clamped to [-6, 6]. Piecewise-uniform levels:
// [0,2] spacing 0.5, [2,4] spacing 1, [4,6] spacing 2; rintf = RNE.
__device__ __forceinline__ float fp4_rne(float x) {
    float ax = fabsf(x);
    float s  = (x < 0.0f) ? -1.0f : 1.0f;
    float q;
    if (ax <= 2.0f)      q = rintf(ax * 2.0f) * 0.5f;
    else if (ax <= 4.0f) q = rintf(ax);
    else                 q = rintf(ax * 0.5f) * 2.0f;
    return s * q;
}

// ---------------------------------------------------------------------------
// Kernel 3: dequantize W (emulated nvfp4) -> f16, one thread per 16-k block.
// Layout of wq: [G][N][K] row-major (k contiguous), matching WMMA B loads.
// ---------------------------------------------------------------------------
__global__ void __launch_bounds__(256)
nvfp4_dequant_kernel(const float* __restrict__ w, const unsigned* __restrict__ amax,
                     _Float16* __restrict__ wq, int nblk, int blkPerGroup) {
    int b = blockIdx.x * 256 + threadIdx.x;
    if (b >= nblk) return;
    int g = b / blkPerGroup;

    const v4f* pv = (const v4f*)(w + (size_t)b * 16);
    v4f x0 = pv[0], x1 = pv[1], x2 = pv[2], x3 = pv[3];

    float v[16];
#pragma unroll
    for (int i = 0; i < 4; ++i) {
        v[i] = x0[i]; v[4 + i] = x1[i]; v[8 + i] = x2[i]; v[12 + i] = x3[i];
    }
    float m = 0.0f;
#pragma unroll
    for (int i = 0; i < 16; ++i) m = fmaxf(m, fabsf(v[i]));

    float gs = __uint_as_float(amax[g]);
    float bs = fminf(fmaxf(m * (1.0f / FP4_MAX) * gs, F8_TINY), F8_MAX);
    bs = f8e4m3_rne(bs);
    float ts = bs / gs;

    v8h lo, hi;
#pragma unroll
    for (int i = 0; i < 16; ++i) {
        float x = fminf(fmaxf(v[i] / ts, -FP4_MAX), FP4_MAX);
        _Float16 h = (_Float16)(fp4_rne(x) * ts);
        if (i < 8) lo[i] = h; else hi[i - 8] = h;
    }
    v8h* dst = (v8h*)(wq + (size_t)b * 16);
    dst[0] = lo;
    dst[1] = hi;
}

// ---------------------------------------------------------------------------
// Kernel 4: convert hidden states f32 -> f16 (8 elements per thread).
// ---------------------------------------------------------------------------
__global__ void __launch_bounds__(256)
nvfp4_cvt_kernel(const float* __restrict__ H, _Float16* __restrict__ Hq,
                 long long n) {
    long long i = ((long long)blockIdx.x * 256 + threadIdx.x) * 8;
    if (i >= n) return;
    v4f a = *(const v4f*)(H + i);
    v4f b = *(const v4f*)(H + i + 4);
    v8h o;
#pragma unroll
    for (int j = 0; j < 4; ++j) {
        o[j]     = (_Float16)a[j];
        o[4 + j] = (_Float16)b[j];
    }
    *(v8h*)(Hq + i) = o;
}

// ---------------------------------------------------------------------------
// Kernel 5: grouped GEMM via V_WMMA_F32_16X16X32_F16.
// Group-aligned M-tiles: blockIdx.y enumerates (group, 16-row tile) pairs, so
// a tile never crosses a group boundary -> no masking in the inner loop.
// Block: 256 threads = 8 wave32s -> 16(M) x 512(N); wave w owns cols
// [w*64, w*64+64) as 4 wmma accumulators over K. Inner k-step:
// 2 f16 A loads + 8 f16 B loads + 4 WMMA, no VALU.
// ---------------------------------------------------------------------------
__global__ void __launch_bounds__(256)
nvfp4_gemm_kernel(const _Float16* __restrict__ Hq, const _Float16* __restrict__ W,
                  const int* __restrict__ offs, float* __restrict__ out,
                  int N, int K, int G) {
    // ---- map blockIdx.y -> (group g, tile start row) ----
    const int mt = blockIdx.y;
    int g = -1, tstart = 0, tend = 0, cum = 0;
    for (int i = 0; i < G; ++i) {
        const int o0 = offs[i], o1 = offs[i + 1];
        const int nt = (o1 - o0 + 15) >> 4;
        if (g < 0 && mt < cum + nt) {
            g = i;
            tstart = o0 + ((mt - cum) << 4);
            tend = o1;
        }
        cum += nt;
    }
    if (g < 0) return;   // beyond total tile count

    const int lane = threadIdx.x & 31;
    const int wave = threadIdx.x >> 5;
    const int n0   = blockIdx.x * 512 + wave * 64;
    const int lm   = lane & 15;
    const int hiL  = lane >> 4;                 // 0 or 1
    const int kb8  = hiL * 16;                  // A: k sub-offset, contiguous k = 16*hiL + s
    const int kb16 = hiL * 16;                  // B: k sub-offset, contiguous k = 16*hiL + s

    // A row for this lane; clamp into group for partial tail tiles
    int trow = tstart + lm;
    if (trow >= tend) trow = tend - 1;

    const _Float16* __restrict__ arow = Hq + (size_t)trow * K + kb8;

    const _Float16* __restrict__ wg  = W + (size_t)g * N * K;
    const _Float16* __restrict__ b0p = wg + (size_t)(n0 +  0 + lm) * K + kb16;
    const _Float16* __restrict__ b1p = wg + (size_t)(n0 + 16 + lm) * K + kb16;
    const _Float16* __restrict__ b2p = wg + (size_t)(n0 + 32 + lm) * K + kb16;
    const _Float16* __restrict__ b3p = wg + (size_t)(n0 + 48 + lm) * K + kb16;

    v8f acc0 = {}, acc1 = {}, acc2 = {}, acc3 = {};

#pragma unroll 4
    for (int k0 = 0; k0 < K; k0 += 32) {
        // ---- A tile (16x32 f16): halfs {kb8..kb8+15} contiguous ----
        v8h al = *(const v8h*)(arow + k0);
        v8h ah = *(const v8h*)(arow + k0 + 8);
        v16h a = __builtin_shufflevector(al, ah, 0,1,2,3,4,5,6,7,8,9,10,11,12,13,14,15);

        // ---- B tiles (32x16 f16): 16 contiguous k per lane ----
        v8h l0 = *(const v8h*)(b0p + k0), h0 = *(const v8h*)(b0p + k0 + 8);
        v8h l1 = *(const v8h*)(b1p + k0), h1 = *(const v8h*)(b1p + k0 + 8);
        v8h l2 = *(const v8h*)(b2p + k0), h2 = *(const v8h*)(b2p + k0 + 8);
        v8h l3 = *(const v8h*)(b3p + k0), h3 = *(const v8h*)(b3p + k0 + 8);
        v16h b0 = __builtin_shufflevector(l0, h0, 0,1,2,3,4,5,6,7,8,9,10,11,12,13,14,15);
        v16h b1 = __builtin_shufflevector(l1, h1, 0,1,2,3,4,5,6,7,8,9,10,11,12,13,14,15);
        v16h b2 = __builtin_shufflevector(l2, h2, 0,1,2,3,4,5,6,7,8,9,10,11,12,13,14,15);
        v16h b3 = __builtin_shufflevector(l3, h3, 0,1,2,3,4,5,6,7,8,9,10,11,12,13,14,15);

        acc0 = __builtin_amdgcn_wmma_f32_16x16x32_f16(false, a, false, b0, (short)0, acc0, false, false);
        acc1 = __builtin_amdgcn_wmma_f32_16x16x32_f16(false, a, false, b1, (short)0, acc1, false, false);
        acc2 = __builtin_amdgcn_wmma_f32_16x16x32_f16(false, a, false, b2, (short)0, acc2, false, false);
        acc3 = __builtin_amdgcn_wmma_f32_16x16x32_f16(false, a, false, b3, (short)0, acc3, false, false);
    }

    // ---- store D: VGPR r -> row (tstart + r + 8*hiL), col = n0 + (lane&15);
    //      guard rows beyond the group end (partial tail tile) ----
    const int rbase = hiL * 8;
#pragma unroll
    for (int r = 0; r < 8; ++r) {
        const int row = tstart + rbase + r;
        if (row < tend) {
            float* orow = out + (size_t)row * N + (n0 + lm);
            orow[0]  = acc0[r];
            orow[16] = acc1[r];
            orow[32] = acc2[r];
            orow[48] = acc3[r];
        }
    }
}

// ---------------------------------------------------------------------------
// Launch
// ---------------------------------------------------------------------------
extern "C" void kernel_launch(void* const* d_in, const int* in_sizes, int n_in,
                              void* d_out, int out_size, void* d_ws, size_t ws_size,
                              hipStream_t stream) {
    const float* H    = (const float*)d_in[0];
    const float* Wf32 = (const float*)d_in[1];
    const int*   offs = (const int*)d_in[2];
    float*       out  = (float*)d_out;

    const int G = in_sizes[2] - 1;                 // 8
    const int K = 1024;                            // reference K
    const long long perGroup = (long long)in_sizes[1] / G;   // N*K
    const int N = (int)(perGroup / K);             // 2048
    const int T = in_sizes[0] / K;                 // 4096

    // workspace layout:
    //   [0, 32)                      : per-group absmax (uint)
    //   [256, 256 + G*N*K*2)         : f16 dequantized weights
    //   [256 + G*N*K*2, ... + T*K*2) : f16 hidden states
    unsigned*  amax = (unsigned*)d_ws;
    _Float16*  Wq   = (_Float16*)((char*)d_ws + 256);
    _Float16*  Hq   = Wq + (size_t)G * N * K;

    nvfp4_init_kernel<<<1, 32, 0, stream>>>(amax, G);

    nvfp4_absmax_kernel<<<dim3(64, G), 256, 0, stream>>>(Wf32, amax, perGroup);

    const int nblk = (int)((long long)G * perGroup / 16);
    const int blkPerGroup = (int)(perGroup / 16);
    nvfp4_dequant_kernel<<<(nblk + 255) / 256, 256, 0, stream>>>(Wf32, amax, Wq,
                                                                 nblk, blkPerGroup);

    const long long nH = (long long)T * K;
    nvfp4_cvt_kernel<<<(int)((nH / 8 + 255) / 256), 256, 0, stream>>>(H, Hq, nH);

    const int mtiles = (T >> 4) + G;               // upper bound on group-aligned tiles
    nvfp4_gemm_kernel<<<dim3(N / 512, mtiles), 256, 0, stream>>>(Hq, Wq, offs, out,
                                                                 N, K, G);
}
